// TimeAwareSelfAttention_24060406792458
// MI455X (gfx1250) — compile-verified, matrix operand order checked by reading, not run on hardware
//
#include <hip/hip_runtime.h>

typedef __attribute__((ext_vector_type(8)))  _Float16 v8h;
typedef __attribute__((ext_vector_type(16))) _Float16 v16h;
typedef __attribute__((ext_vector_type(8)))  float    v8f;

#define WMMA_F32_F16(a, b, c) \
    __builtin_amdgcn_wmma_f32_16x16x32_f16(false, (a), false, (b), (short)0, (c), false, false)

// ---------------------------------------------------------------------------
// Fragment loader for both A-operand (16xK rows of the left matrix) and the
// B-operand (rows of B^T, i.e. rows of a row-major NxK matrix).
// ISA 16-bit A layout (CDNA5 ISA 7.12.2):
//   lanes 0-15  : M = lane,     elems 0..7 = K 0..7,   elems 8..15 = K 16..23
//   lanes 16-31 : M = lane-16,  elems 0..7 = K 8..15,  elems 8..15 = K 24..31
// ---------------------------------------------------------------------------
__device__ __forceinline__ v16h load_frag(const _Float16* __restrict__ base,
                                          int ld, int row0, int k0) {
    int lane = threadIdx.x & 31;
    int r    = lane & 15;
    int hg   = lane >> 4;
    const _Float16* p = base + (size_t)(row0 + r) * (size_t)ld + (size_t)(k0 + hg * 8);
    v8h lo = *(const v8h*)p;          // K = k0 + hg*8 .. +7
    v8h hi = *(const v8h*)(p + 16);   // K = k0 + 16 + hg*8 .. +7
    return __builtin_shufflevector(lo, hi, 0,1,2,3,4,5,6,7,8,9,10,11,12,13,14,15);
}

__device__ __forceinline__ v16h load_frag_lds(const _Float16* base /*16 x 40 halfs*/) {
    int lane = threadIdx.x & 31;
    int r    = lane & 15;
    int hg   = lane >> 4;
    const _Float16* p = base + r * 40 + hg * 8;
    v8h lo = *(const v8h*)p;
    v8h hi = *(const v8h*)(p + 16);
    return __builtin_shufflevector(lo, hi, 0,1,2,3,4,5,6,7,8,9,10,11,12,13,14,15);
}

// ---------------------------------------------------------------------------
// fp32 -> f16 conversion
// ---------------------------------------------------------------------------
__global__ __launch_bounds__(256) void cvt_f32_f16(const float* __restrict__ in,
                                                   _Float16* __restrict__ out, int n) {
    int i = blockIdx.x * 256 + threadIdx.x;
    if (i < n) out[i] = (_Float16)in[i];
}

// ---------------------------------------------------------------------------
// WMMA GEMM:  D[m][n] = sum_k A[m][k] * Bw[n][k] + bias[n]
//   A : M x K row-major f16, Bw : N x K row-major f16 (torch Linear weight)
//   mode 0: store f16 row-major [M][N]
//   mode 1: store f16 transposed [b][n][token]   (b = m>>10, token = m&1023)
//   mode 2: store f32 row-major [M][N]
// Block: 128 threads = 4 waves (2x2), WG tile 128x128, wave tile 64x64:
// 16 accumulators, 8 fragment loads : 16 WMMAs per k-step (1 load / WMMA).
// ---------------------------------------------------------------------------
__global__ __launch_bounds__(128) void gemm_wmma(const _Float16* __restrict__ A,
                                                 const _Float16* __restrict__ Bw,
                                                 const float* __restrict__ bias,
                                                 void* __restrict__ out,
                                                 int M, int N, int K, int mode) {
    int w    = threadIdx.x >> 5;
    int lane = threadIdx.x & 31;
    int m0 = blockIdx.y * 128 + (w >> 1) * 64;
    int n0 = blockIdx.x * 128 + (w & 1) * 64;

    v8f acc[4][4] = {};
    for (int k0 = 0; k0 < K; k0 += 32) {
        if (k0 + 64 < K) {  // hint the next-next k-tile toward the caches
            __builtin_prefetch(A  + (size_t)m0 * K + k0 + 64, 0, 1);
            __builtin_prefetch(Bw + (size_t)n0 * K + k0 + 64, 0, 1);
        }
        v16h a0 = load_frag(A,  K, m0,      k0);
        v16h a1 = load_frag(A,  K, m0 + 16, k0);
        v16h a2 = load_frag(A,  K, m0 + 32, k0);
        v16h a3 = load_frag(A,  K, m0 + 48, k0);
        v16h b0 = load_frag(Bw, K, n0,      k0);
        v16h b1 = load_frag(Bw, K, n0 + 16, k0);
        v16h b2 = load_frag(Bw, K, n0 + 32, k0);
        v16h b3 = load_frag(Bw, K, n0 + 48, k0);
        acc[0][0] = WMMA_F32_F16(a0, b0, acc[0][0]);
        acc[0][1] = WMMA_F32_F16(a0, b1, acc[0][1]);
        acc[0][2] = WMMA_F32_F16(a0, b2, acc[0][2]);
        acc[0][3] = WMMA_F32_F16(a0, b3, acc[0][3]);
        acc[1][0] = WMMA_F32_F16(a1, b0, acc[1][0]);
        acc[1][1] = WMMA_F32_F16(a1, b1, acc[1][1]);
        acc[1][2] = WMMA_F32_F16(a1, b2, acc[1][2]);
        acc[1][3] = WMMA_F32_F16(a1, b3, acc[1][3]);
        acc[2][0] = WMMA_F32_F16(a2, b0, acc[2][0]);
        acc[2][1] = WMMA_F32_F16(a2, b1, acc[2][1]);
        acc[2][2] = WMMA_F32_F16(a2, b2, acc[2][2]);
        acc[2][3] = WMMA_F32_F16(a2, b3, acc[2][3]);
        acc[3][0] = WMMA_F32_F16(a3, b0, acc[3][0]);
        acc[3][1] = WMMA_F32_F16(a3, b1, acc[3][1]);
        acc[3][2] = WMMA_F32_F16(a3, b2, acc[3][2]);
        acc[3][3] = WMMA_F32_F16(a3, b3, acc[3][3]);
    }

    int r  = lane & 15;
    int hg = lane >> 4;
#pragma unroll
    for (int i = 0; i < 4; ++i)
#pragma unroll
        for (int j = 0; j < 4; ++j) {
            int nn = n0 + 16 * j + r;
            float bv = bias[nn];
#pragma unroll
            for (int rr = 0; rr < 8; ++rr) {
                int row = m0 + 16 * i + rr + 8 * hg;
                float v = acc[i][j][rr] + bv;
                if (mode == 2) {
                    ((float*)out)[(size_t)row * N + nn] = v;
                } else if (mode == 1) {
                    int bb = row >> 10, tok = row & 1023;
                    ((_Float16*)out)[((size_t)(bb * 1024 + nn)) * 1024 + tok] = (_Float16)v;
                } else {
                    ((_Float16*)out)[(size_t)row * N + nn] = (_Float16)v;
                }
            }
        }
}

// ---------------------------------------------------------------------------
// Flash attention with time-bucket bias + causal mask.
// One wave handles 16 query rows of one (b, h). 8 waves / block share (b, h).
// ---------------------------------------------------------------------------
__global__ __launch_bounds__(256) void attn_wmma(const _Float16* __restrict__ Q16,
                                                 const _Float16* __restrict__ K16,
                                                 const _Float16* __restrict__ Vt16,
                                                 const float* __restrict__ ts,
                                                 const float* __restrict__ tb,
                                                 _Float16* __restrict__ O16) {
    __shared__ float tb_s[32];
    __shared__ __align__(16) _Float16 plds[8][16][40];

    int wl   = threadIdx.x >> 5;
    int lane = threadIdx.x & 31;
    int wfirst = blockIdx.x * 8;
    int h = (wfirst >> 6) & 15;                 // identical for all 8 waves
    if (threadIdx.x < 32) tb_s[threadIdx.x] = tb[threadIdx.x * 16 + h];
    __syncthreads();

    int w  = wfirst + wl;
    int qt = w & 63;
    int b  = w >> 10;
    int q0 = qt * 16;
    int r  = lane & 15;
    int hg = lane >> 4;

    const _Float16* qbase = Q16 + ((size_t)(b * 1024 + q0)) * 1024 + h * 64;
    v16h qa0 = load_frag(qbase, 1024, 0, 0);
    v16h qa1 = load_frag(qbase, 1024, 0, 32);

    float tsq[8];
#pragma unroll
    for (int rr = 0; rr < 8; ++rr)
        tsq[rr] = ts[b * 1024 + q0 + rr + 8 * hg];

    float mi[8], li[8];
#pragma unroll
    for (int rr = 0; rr < 8; ++rr) { mi[rr] = -3.0e38f; li[rr] = 0.0f; }
    v8f o0 = {}, o1 = {}, o2 = {}, o3 = {};

    const float scale    = 0.125f;                         // 1/sqrt(64)
    const float inv_step = 30.0f / __log2f(2592000.0f);    // (NB-2)/log2(MAX_DT)
    const _Float16* vtb  = Vt16 + ((size_t)(b * 1024 + h * 64)) * 1024;

    for (int jb = 0; jb <= q0 + 15; jb += 32) {
        const _Float16* kbase = K16 + ((size_t)(b * 1024 + jb)) * 1024 + h * 64;
        v16h kf00 = load_frag(kbase, 1024, 0,  0);
        v16h kf01 = load_frag(kbase, 1024, 0,  32);
        v16h kf10 = load_frag(kbase, 1024, 16, 0);
        v16h kf11 = load_frag(kbase, 1024, 16, 32);

        v8f s0 = {}, s1 = {};
        s0 = WMMA_F32_F16(qa0, kf00, s0);
        s0 = WMMA_F32_F16(qa1, kf01, s0);
        s1 = WMMA_F32_F16(qa0, kf10, s1);
        s1 = WMMA_F32_F16(qa1, kf11, s1);

        float tsk0 = ts[b * 1024 + jb + r];
        float tsk1 = ts[b * 1024 + jb + 16 + r];

#pragma unroll
        for (int rr = 0; rr < 8; ++rr) {
            int qi = q0 + rr + 8 * hg;
            // --- time-bucket bias (faithful port of _bucket) ---
            float dt0 = fmaxf(fabsf(tsq[rr] - tsk0), 1.0f);
            float dt1 = fmaxf(fabsf(tsq[rr] - tsk1), 1.0f);
            int i0 = (int)(__log2f(dt0) * inv_step) + 1;
            int i1 = (int)(__log2f(dt1) * inv_step) + 1;
            i0 = min(max(i0, 1), 30);
            i1 = min(max(i1, 1), 30);
            if (dt0 <= 1.0f) i0 = 0;
            if (dt1 <= 1.0f) i1 = 0;
            if (dt0 >= 2592000.0f) i0 = 31;
            if (dt1 >= 2592000.0f) i1 = 31;

            float sv0 = ((jb + r)      <= qi) ? (s0[rr] * scale + tb_s[i0]) : -10000.0f;
            float sv1 = ((jb + 16 + r) <= qi) ? (s1[rr] * scale + tb_s[i1]) : -10000.0f;

            // --- online softmax: reduce over the 16-lane column group ---
            float mx = fmaxf(sv0, sv1);
            mx = fmaxf(mx, __shfl_xor(mx, 1, 32));
            mx = fmaxf(mx, __shfl_xor(mx, 2, 32));
            mx = fmaxf(mx, __shfl_xor(mx, 4, 32));
            mx = fmaxf(mx, __shfl_xor(mx, 8, 32));
            float mnew  = fmaxf(mi[rr], mx);
            float alpha = __expf(mi[rr] - mnew);
            mi[rr] = mnew;
            float p0 = __expf(sv0 - mnew);
            float p1 = __expf(sv1 - mnew);
            float ps = p0 + p1;
            ps += __shfl_xor(ps, 1, 32);
            ps += __shfl_xor(ps, 2, 32);
            ps += __shfl_xor(ps, 4, 32);
            ps += __shfl_xor(ps, 8, 32);
            li[rr] = li[rr] * alpha + ps;
            o0[rr] *= alpha; o1[rr] *= alpha; o2[rr] *= alpha; o3[rr] *= alpha;

            // stash P in LDS in row-major [m][k] for A-fragment reload
            plds[wl][rr + 8 * hg][r]      = (_Float16)p0;
            plds[wl][rr + 8 * hg][16 + r] = (_Float16)p1;
        }

        // P (16x32) as A-operand; Vt rows (d, contiguous keys) as B-operand
        v16h pa  = load_frag_lds(&plds[wl][0][0]);
        v16h vb0 = load_frag(vtb, 1024, 0,  jb);
        v16h vb1 = load_frag(vtb, 1024, 16, jb);
        v16h vb2 = load_frag(vtb, 1024, 32, jb);
        v16h vb3 = load_frag(vtb, 1024, 48, jb);
        o0 = WMMA_F32_F16(pa, vb0, o0);
        o1 = WMMA_F32_F16(pa, vb1, o1);
        o2 = WMMA_F32_F16(pa, vb2, o2);
        o3 = WMMA_F32_F16(pa, vb3, o3);
    }

    // normalize and write O (f16, [b][token][h*64+d])
    _Float16* ob = O16 + ((size_t)(b * 1024 + q0)) * 1024 + h * 64;
#pragma unroll
    for (int rr = 0; rr < 8; ++rr) {
        float inv = 1.0f / li[rr];
        int row = rr + 8 * hg;
        ob[(size_t)row * 1024 +  0 + r] = (_Float16)(o0[rr] * inv);
        ob[(size_t)row * 1024 + 16 + r] = (_Float16)(o1[rr] * inv);
        ob[(size_t)row * 1024 + 32 + r] = (_Float16)(o2[rr] * inv);
        ob[(size_t)row * 1024 + 48 + r] = (_Float16)(o3[rr] * inv);
    }
}

// ---------------------------------------------------------------------------
// Host-side launcher
// ---------------------------------------------------------------------------
extern "C" void kernel_launch(void* const* d_in, const int* in_sizes, int n_in,
                              void* d_out, int out_size, void* d_ws, size_t ws_size,
                              hipStream_t stream) {
    const float* x  = (const float*)d_in[0];
    const float* ts = (const float*)d_in[1];
    const float* Wq = (const float*)d_in[2];
    const float* bq = (const float*)d_in[3];
    const float* Wk = (const float*)d_in[4];
    const float* bk = (const float*)d_in[5];
    const float* Wv = (const float*)d_in[6];
    const float* bv = (const float*)d_in[7];
    const float* tb = (const float*)d_in[8];
    const float* Wo = (const float*)d_in[9];
    const float* bo = (const float*)d_in[10];
    float* out = (float*)d_out;

    const size_t MN = (size_t)4096 * 1024;   // 4M
    const size_t WN = (size_t)1024 * 1024;   // 1M
    _Float16* ws   = (_Float16*)d_ws;
    _Float16* x16  = ws;
    _Float16* Wq16 = x16 + MN;
    _Float16* Wk16 = Wq16 + WN;
    _Float16* Wv16 = Wk16 + WN;
    _Float16* Wo16 = Wv16 + WN;
    _Float16* Q16  = Wo16 + WN;
    _Float16* K16  = Q16 + MN;
    _Float16* Vt16 = K16 + MN;
    _Float16* O16  = Vt16 + MN;

    cvt_f32_f16<<<(int)((MN + 255) / 256), 256, 0, stream>>>(x,  x16,  (int)MN);
    cvt_f32_f16<<<(int)((WN + 255) / 256), 256, 0, stream>>>(Wq, Wq16, (int)WN);
    cvt_f32_f16<<<(int)((WN + 255) / 256), 256, 0, stream>>>(Wk, Wk16, (int)WN);
    cvt_f32_f16<<<(int)((WN + 255) / 256), 256, 0, stream>>>(Wv, Wv16, (int)WN);
    cvt_f32_f16<<<(int)((WN + 255) / 256), 256, 0, stream>>>(Wo, Wo16, (int)WN);

    dim3 g(8, 32), blk(128);
    gemm_wmma<<<g, blk, 0, stream>>>(x16, Wq16, bq, Q16,  4096, 1024, 1024, 0);
    gemm_wmma<<<g, blk, 0, stream>>>(x16, Wk16, bk, K16,  4096, 1024, 1024, 0);
    gemm_wmma<<<g, blk, 0, stream>>>(x16, Wv16, bv, Vt16, 4096, 1024, 1024, 1);

    attn_wmma<<<512, 256, 0, stream>>>(Q16, K16, Vt16, ts, tb, O16);

    gemm_wmma<<<g, blk, 0, stream>>>(O16, Wo16, bo, out,  4096, 1024, 1024, 2);
}
